// GNNModel_76665166233740
// MI455X (gfx1250) — matrix operand order repaired and checked by reading, not compiled, not run on hardware
//
#include <hip/hip_runtime.h>
#include <hip/hip_bf16.h>

typedef __attribute__((ext_vector_type(16))) _Float16 v16h;
typedef __attribute__((ext_vector_type(8)))  float    v8f;

// A-fragment intra-32 K permutation: stored order [0-7, 16-23, 8-15, 24-31]
__device__ __forceinline__ int apos(int k) {
    int t = (k >> 3) & 3;
    return (t == 1 || t == 2) ? (k ^ 24) : k;
}

// ---------------------------------------------------------------- utilities

__global__ void k_deg(const int* __restrict__ dst, const float* __restrict__ ew,
                      float* __restrict__ deg, int E) {
    int e = blockIdx.x * blockDim.x + threadIdx.x;
    if (e < E) atomicAdd(&deg[dst[e]], ew[e]);
}

__global__ void k_rsqrt(float* __restrict__ d, int n) {
    int i = blockIdx.x * blockDim.x + threadIdx.x;
    if (i < n) d[i] = rsqrtf(d[i] + 1.0f);   // deg includes self-loop weight 1
}

__global__ void k_norm(const int* __restrict__ src, const int* __restrict__ dst,
                       const float* __restrict__ ew, const float* __restrict__ dinv,
                       float* __restrict__ norm, int E) {
    int e = blockIdx.x * blockDim.x + threadIdx.x;
    if (e < E) norm[e] = dinv[src[e]] * ew[e] * dinv[dst[e]];
}

// f32 -> f16 with intra-32-feature permute into WMMA-A friendly order
__global__ void k_pack_x(const float* __restrict__ in, _Float16* __restrict__ Xp,
                         int logF, long long NF) {
    long long idx = (long long)blockIdx.x * blockDim.x + threadIdx.x;
    if (idx >= NF) return;
    int f = (int)(idx & ((1 << logF) - 1));
    long long base = idx - f;
    Xp[base + (f & ~31) + apos(f & 31)] = (_Float16)in[idx];
}

// f32 W[K,F] -> f16 packed WMMA-B fragments:
// Wp[((tn*nks + ks)*32 + lane)*16 + j] = W[(ks*32 + (j>>3)*16 + (lane>>4)*8 + (j&7))*F + tn*16 + (lane&15)]
__global__ void k_pack_w(const float* __restrict__ W, _Float16* __restrict__ Wp,
                         int nks, int F, int total) {
    int p = blockIdx.x * blockDim.x + threadIdx.x;
    if (p >= total) return;
    int j    = p & 15;
    int lane = (p >> 4) & 31;
    int rest = p >> 9;
    int ks   = rest % nks;
    int tn   = rest / nks;
    int k    = ks * 32 + (j >> 3) * 16 + (lane >> 4) * 8 + (j & 7);
    int col  = tn * 16 + (lane & 15);
    Wp[p] = (_Float16)W[k * F + col];
}

// ------------------------------------------------- WMMA GEMM  H = X @ W
// One wave -> 32x32 f32 output (2x2 tiles of 16x16), K-loop in steps of 32.
// A and B fragments are single contiguous 32B loads (pre-packed layouts).
__global__ __launch_bounds__(128)
void k_gemm_wmma(const _Float16* __restrict__ Xp,  // [N, K] A-permuted f16
                 const _Float16* __restrict__ Wp,  // packed B fragments
                 float* __restrict__ H,            // [N, F] f32
                 int N, int K, int F) {
    const int lane = threadIdx.x & 31;
    const int wave = threadIdx.x >> 5;
    const int lh   = lane & 15;
    const int hi   = lane >> 4;

    const int row0 = blockIdx.x * 32;
    const int col0 = (blockIdx.y * 4 + wave) * 32;
    if (col0 >= F) return;

    const int nks = K >> 5;
    int ar0 = row0 + lh;       if (ar0 >= N) ar0 = N - 1;
    int ar1 = row0 + 16 + lh;  if (ar1 >= N) ar1 = N - 1;
    const _Float16* pa0 = Xp + (long long)ar0 * K + hi * 16;
    const _Float16* pa1 = Xp + (long long)ar1 * K + hi * 16;
    const int tn = col0 >> 4;
    const _Float16* pb0 = Wp + ((long long)(tn       * nks) * 32 + lane) * 16;
    const _Float16* pb1 = Wp + ((long long)((tn + 1) * nks) * 32 + lane) * 16;

    v8f c00 = {}, c01 = {}, c10 = {}, c11 = {};
    for (int ks = 0; ks < nks; ++ks) {
        v16h a0 = *(const v16h*)(pa0 + ks * 32);
        v16h a1 = *(const v16h*)(pa1 + ks * 32);
        v16h b0 = *(const v16h*)(pb0 + (long long)ks * 512);
        v16h b1 = *(const v16h*)(pb1 + (long long)ks * 512);
        c00 = __builtin_amdgcn_wmma_f32_16x16x32_f16(false, a0, false, b0, (short)0, c00, false, false);
        c01 = __builtin_amdgcn_wmma_f32_16x16x32_f16(false, a0, false, b1, (short)0, c01, false, false);
        c10 = __builtin_amdgcn_wmma_f32_16x16x32_f16(false, a1, false, b0, (short)0, c10, false, false);
        c11 = __builtin_amdgcn_wmma_f32_16x16x32_f16(false, a1, false, b1, (short)0, c11, false, false);
    }

#pragma unroll
    for (int r = 0; r < 8; ++r) {
        int m0 = row0 + r + hi * 8;
        int m1 = m0 + 16;
        int cA = col0 + lh;
        if (m0 < N) {
            H[(long long)m0 * F + cA]      = c00[r];
            H[(long long)m0 * F + cA + 16] = c01[r];
        }
        if (m1 < N) {
            H[(long long)m1 * F + cA]      = c10[r];
            H[(long long)m1 * F + cA + 16] = c11[r];
        }
    }
}

// ----------------------------------------------- self-loop + bias init
__global__ void k_selfbias(const float* __restrict__ H, const float* __restrict__ dinv,
                           const float* __restrict__ b, float* __restrict__ out,
                           int logF, long long NF) {
    long long idx = (long long)blockIdx.x * blockDim.x + threadIdx.x;
    if (idx >= NF) return;
    int i = (int)(idx >> logF);
    int f = (int)(idx & ((1 << logF) - 1));
    float di = dinv[i];
    out[idx] = H[idx] * di * di + b[f];
}

// ----------------------------------------------- edge scatter-add
// out[dst] += H[src] * norm[e]  (edge scalars wave-broadcast; gather/atomics coalesced)
__global__ void k_agg(const int* __restrict__ src, const int* __restrict__ dst,
                      const float* __restrict__ norm, const float* __restrict__ H,
                      float* __restrict__ out, int logF, long long EF) {
    long long idx = (long long)blockIdx.x * blockDim.x + threadIdx.x;
    if (idx >= EF) return;
    int e = (int)(idx >> logF);
    int f = (int)(idx & ((1 << logF) - 1));
    int s = src[e], d = dst[e];
    float v = H[((long long)s << logF) + f] * norm[e];
    atomicAdd(&out[((long long)d << logF) + f], v);
}

// ----------------------------------------------- relu (in-place) + packed f16 copy
__global__ void k_relu_cvt(float* __restrict__ out, _Float16* __restrict__ Xp,
                           int logF, long long NF) {
    long long idx = (long long)blockIdx.x * blockDim.x + threadIdx.x;
    if (idx >= NF) return;
    float v = fmaxf(out[idx], 0.0f);
    out[idx] = v;
    int f = (int)(idx & ((1 << logF) - 1));
    long long base = idx - f;
    Xp[base + (f & ~31) + apos(f & 31)] = (_Float16)v;
}

// ----------------------------------------------- global max pool (F == 256)
// Post-ReLU values >= 0, so int-bit atomicMax == IEEE float max.
__global__ void k_maxpool(const float* __restrict__ h3, int* __restrict__ gbits,
                          int N) {
    int f  = threadIdx.x;              // 0..255 column
    int r0 = blockIdx.x * 256;
    int r1 = min(r0 + 256, N);
    float m = 0.0f;
    for (int r = r0; r < r1; ++r)
        m = fmaxf(m, h3[(long long)r * 256 + f]);
    atomicMax(&gbits[f], __float_as_int(m));
}

// ----------------------------------------------- head MLP + log_softmax
__global__ void k_mlp(const float* __restrict__ g,
                      const float* __restrict__ Wl1, const float* __restrict__ bl1,
                      const float* __restrict__ Wl2, const float* __restrict__ bl2,
                      float* __restrict__ out) {
    __shared__ float sg[256];
    __shared__ float h1[128];
    __shared__ float lg[10];
    int t = threadIdx.x;
    sg[t] = g[t];
    __syncthreads();
    if (t < 128) {
        float acc = bl1[t];
        for (int k = 0; k < 256; ++k) acc = fmaf(sg[k], Wl1[k * 128 + t], acc);
        h1[t] = fmaxf(acc, 0.0f);
    }
    __syncthreads();
    if (t < 10) {
        float acc = bl2[t];
        for (int k = 0; k < 128; ++k) acc = fmaf(h1[k], Wl2[k * 10 + t], acc);
        lg[t] = acc;
    }
    __syncthreads();
    if (t == 0) {
        float m = lg[0];
        for (int i = 1; i < 10; ++i) m = fmaxf(m, lg[i]);
        float s = 0.0f;
        for (int i = 0; i < 10; ++i) s += expf(lg[i] - m);
        float lse = m + logf(s);
        for (int i = 0; i < 10; ++i) out[i] = lg[i] - lse;
    }
}

// ================================================================ launcher

extern "C" void kernel_launch(void* const* d_in, const int* in_sizes, int n_in,
                              void* d_out, int out_size, void* d_ws, size_t ws_size,
                              hipStream_t stream) {
    const float* x   = (const float*)d_in[0];
    const int*   ei  = (const int*)d_in[1];
    const float* ew  = (const float*)d_in[2];
    const float* W[3] = {(const float*)d_in[3], (const float*)d_in[5], (const float*)d_in[7]};
    const float* B[3] = {(const float*)d_in[4], (const float*)d_in[6], (const float*)d_in[8]};
    const float* Wl1 = (const float*)d_in[9];
    const float* bl1 = (const float*)d_in[10];
    const float* Wl2 = (const float*)d_in[11];
    const float* bl2 = (const float*)d_in[12];

    const int N = in_sizes[0] / 128;
    const int E = in_sizes[2];
    const int* src = ei;
    const int* dst = ei + E;

    // ---- workspace carve-out
    char* ws = (char*)d_ws;
    auto carve = [&](size_t bytes) {
        char* p = ws;
        ws += (bytes + 255) & ~(size_t)255;
        return (void*)p;
    };
    float*    dinv = (float*)   carve((size_t)N * 4);          // deg -> dinv in place
    float*    nrm  = (float*)   carve((size_t)E * 4);
    _Float16* Xp   = (_Float16*)carve((size_t)N * 256 * 2);    // A-packed activations
    _Float16* Wp   = (_Float16*)carve((size_t)128 * 256 * 2);  // B-packed weights
    float*    H    = (float*)   carve((size_t)N * 256 * 4);
    float*    OUTb = (float*)   carve((size_t)N * 256 * 4);
    int*      g    = (int*)     carve(256 * 4);

    const int TB = 256;
    auto nblk = [&](long long n) { return (unsigned)((n + TB - 1) / TB); };

    // ---- degrees / normalization (edge-invariant across layers)
    hipMemsetAsync(dinv, 0, (size_t)N * 4, stream);
    k_deg  <<<nblk(E), TB, 0, stream>>>(dst, ew, dinv, E);
    k_rsqrt<<<nblk(N), TB, 0, stream>>>(dinv, N);
    k_norm <<<nblk(E), TB, 0, stream>>>(src, dst, ew, dinv, nrm, E);

    // ---- packed f16 copy of input activations (F = 128, logF = 7)
    k_pack_x<<<nblk((long long)N * 128), TB, 0, stream>>>(x, Xp, 7, (long long)N * 128);

    const int Fin [3] = {128,  64, 128};
    const int Fout[3] = { 64, 128, 256};
    const int logF[3] = {  6,   7,   8};

    for (int l = 0; l < 3; ++l) {
        const int K = Fin[l], F = Fout[l];
        const int nks = K >> 5;
        k_pack_w<<<nblk((long long)K * F), TB, 0, stream>>>(W[l], Wp, nks, F, K * F);

        dim3 ggrid((N + 31) / 32, (F + 127) / 128);
        k_gemm_wmma<<<ggrid, 128, 0, stream>>>(Xp, Wp, H, N, K, F);

        long long NF = (long long)N * F;
        k_selfbias<<<nblk(NF), TB, 0, stream>>>(H, dinv, B[l], OUTb, logF[l], NF);

        long long EF = (long long)E << logF[l];
        k_agg<<<nblk(EF), TB, 0, stream>>>(src, dst, nrm, H, OUTb, logF[l], EF);

        k_relu_cvt<<<nblk(NF), TB, 0, stream>>>(OUTb, Xp, logF[l], NF);
    }

    // ---- global max pool over nodes (F = 256)
    hipMemsetAsync(g, 0, 256 * 4, stream);
    k_maxpool<<<(N + 255) / 256, 256, 0, stream>>>(OUTb, g, N);

    // ---- head MLP + log_softmax -> d_out[10]
    k_mlp<<<1, 256, 0, stream>>>((const float*)g, Wl1, bl1, Wl2, bl2, (float*)d_out);
}
